// causalMultiHeadSelfAttention_10849087389820
// MI455X (gfx1250) — matrix operand order repaired and checked by reading, not compiled, not run on hardware
//
#include <hip/hip_runtime.h>
#include <math.h>

// MI455X / gfx1250, wave32. All matmuls via v_wmma_f32_16x16x32_f16.
typedef __attribute__((ext_vector_type(16))) _Float16 v16h;
typedef __attribute__((ext_vector_type(8)))  _Float16 v8h;
typedef __attribute__((ext_vector_type(8)))  float    v8f;
typedef __attribute__((ext_vector_type(4)))  int      v4i;

#define WMMA_F16(a, b, c) \
  __builtin_amdgcn_wmma_f32_16x16x32_f16(false, (a), false, (b), (short)0, (c), false, false)

// ---- gfx1250 async global->LDS path (guarded; sync fallback) -------------
// Probe result (round 2): builtin exists; param0 = AS1 (global) v4i*,
// param1 = LDS v4i*, then two int immediates (offset, cpol).
#if defined(__HIP_DEVICE_COMPILE__) &&                                   \
    __has_builtin(__builtin_amdgcn_global_load_async_to_lds_b128) &&     \
    __has_builtin(__builtin_amdgcn_s_wait_asynccnt)
#define HAVE_ASYNC 1
#else
#define HAVE_ASYNC 0
#endif

#if HAVE_ASYNC
#define GAS __attribute__((address_space(1)))
#define LAS __attribute__((address_space(3)))
__device__ __forceinline__ void async_cp16(void* lds, const void* g) {
  __builtin_amdgcn_global_load_async_to_lds_b128((GAS v4i*)g, (LAS v4i*)lds, 0, 0);
}
#define ASYNC_WAIT() __builtin_amdgcn_s_wait_asynccnt(0)
#else
#define ASYNC_WAIT() ((void)0)
#endif

constexpr int D_MODEL = 1024;
constexpr int NH      = 16;
constexpr int DK      = 64;
constexpr int SEQ     = 2048;
constexpr int BATCH   = 2;
constexpr int BS      = BATCH * SEQ;      // 4096 rows
constexpr int NC      = 3 * D_MODEL;      // combined q|k|v output columns

// ---- WMMA fragment loaders from LDS (f16) -------------------------------
// A (16x32, MxK): lane<16 holds row M=lane, K {0..7,16..23}; lane>=16 same
// rows, K {8..15,24..31}.
__device__ __forceinline__ v16h ld_a_frag(const _Float16* base, int stride,
                                          int row0, int kbase) {
  const int lane = threadIdx.x & 31;
  const _Float16* p = base + (row0 + (lane & 15)) * stride + kbase + ((lane >> 4) << 3);
  v8h lo = *(const v8h*)(p);
  v8h hi = *(const v8h*)(p + 16);
  v16h v;
#pragma unroll
  for (int i = 0; i < 8; ++i) { v[i] = lo[i]; v[i + 8] = hi[i]; }
  return v;
}

// B (32x16, KxN): lane&15 selects column N; lanes 0-15 hold K {0..7,16..23},
// lanes 16-31 hold K {8..15,24..31} (same order as A).
__device__ __forceinline__ v16h ld_b_frag(const _Float16* base, int stride,
                                          int row0, int kbase) {
  const int lane = threadIdx.x & 31;
  const _Float16* p = base + (row0 + (lane & 15)) * stride + kbase + ((lane >> 4) << 3);
  v8h lo = *(const v8h*)(p);
  v8h hi = *(const v8h*)(p + 16);
  v16h v;
#pragma unroll
  for (int i = 0; i < 8; ++i) { v[i] = lo[i]; v[i + 8] = hi[i]; }
  return v;
}

__device__ __forceinline__ v8h cvt8(const float* src) {
  float4 f0 = *(const float4*)src;
  float4 f1 = *(const float4*)(src + 4);
  v8h h;
  h[0] = (_Float16)f0.x; h[1] = (_Float16)f0.y; h[2] = (_Float16)f0.z; h[3] = (_Float16)f0.w;
  h[4] = (_Float16)f1.x; h[5] = (_Float16)f1.y; h[6] = (_Float16)f1.z; h[7] = (_Float16)f1.w;
  return h;
}

// =========================================================================
// Kernel 1: fused QKV projection + RoPE.  out[gm, c] = sum_d x[gm,d]*w[c,d]
// Block tile 128x64, K-slab 64; 8 waves, each wave computes 32x32 (2x2 accs,
// 8 WMMAs per K-slab).  Q gets RoPE + 1/8 scale, K gets RoPE, V passthrough.
// Stored f16 as [B][H][S][DK].  Grid (BS/128, NC/64), block 256.
// =========================================================================
__global__ __launch_bounds__(256) void qkv_rope_kernel(
    const float* __restrict__ x,  const float* __restrict__ wq,
    const float* __restrict__ wk, const float* __restrict__ wv,
    const int* __restrict__ tok,
    _Float16* __restrict__ Qh, _Float16* __restrict__ Kh,
    _Float16* __restrict__ Vh) {
  __shared__ _Float16 xs[128 * 72];   // x tile   (rows 128 x 64 K, stride 72)
  __shared__ _Float16 bsm[64 * 72];   // w tile   (rows 64  x 64 K)

  const int tid  = threadIdx.x;
  const int lane = tid & 31;
  const int wave = tid >> 5;
  const int m0   = blockIdx.x * 128;
  const int n0   = blockIdx.y * 64;        // combined column base
  const int which = n0 >> 10;              // 0:q 1:k 2:v (uniform per block)
  const int nl0   = n0 & (D_MODEL - 1);
  const float* wsel = (which == 0) ? wq : (which == 1) ? wk : wv;

  const int msub = (wave & 3) << 5;        // 0,32,64,96
  const int nsub = (wave >> 2) << 5;       // 0,32

  const int xrow = tid >> 1, xcb = (tid & 1) << 5;   // x tile: 32 halfs/thread
  const int wrow = tid >> 2, wcb = (tid & 3) << 4;   // w tile: 16 halfs/thread

  v8f acc[2][2] = {};

  for (int kk = 0; kk < D_MODEL; kk += 64) {
    __syncthreads();
    {
      const float* sx = x + (size_t)(m0 + xrow) * D_MODEL + kk + xcb;
#pragma unroll
      for (int q = 0; q < 4; ++q)
        *(v8h*)(xs + xrow * 72 + xcb + q * 8) = cvt8(sx + q * 8);
    }
    {
      const float* sw = wsel + (size_t)(nl0 + wrow) * D_MODEL + kk + wcb;
#pragma unroll
      for (int q = 0; q < 2; ++q)
        *(v8h*)(bsm + wrow * 72 + wcb + q * 8) = cvt8(sw + q * 8);
    }
    __syncthreads();

#pragma unroll
    for (int ks = 0; ks < 64; ks += 32) {
      v16h a0 = ld_a_frag(xs, 72, msub, ks);
      v16h a1 = ld_a_frag(xs, 72, msub + 16, ks);
      v16h b0 = ld_b_frag(bsm, 72, nsub, ks);
      v16h b1 = ld_b_frag(bsm, 72, nsub + 16, ks);
      acc[0][0] = WMMA_F16(a0, b0, acc[0][0]);
      acc[0][1] = WMMA_F16(a0, b1, acc[0][1]);
      acc[1][0] = WMMA_F16(a1, b0, acc[1][0]);
      acc[1][1] = WMMA_F16(a1, b1, acc[1][1]);
    }
  }

  // Epilogue: RoPE (q,k) + store f16 into [B][H][S][DK]
  const int hi = lane >> 4, ln = lane & 15;
  _Float16* dst = (which == 0) ? Qh : (which == 1) ? Kh : Vh;
#pragma unroll
  for (int mi = 0; mi < 2; ++mi) {
#pragma unroll
    for (int nb = 0; nb < 2; ++nb) {
      v8f a = acc[mi][nb];
      const int c    = n0 + nsub + nb * 16 + ln;
      const int head = (c & (D_MODEL - 1)) >> 6;
      const int d    = c & (DK - 1);
      // inv_freq = theta^(-(d/2)/32) = 2^(-(d/2) * log2(1e4)/32)
      const float inv = exp2f(-(float)(d >> 1) * 0.41524101186092557f);
#pragma unroll
      for (int r = 0; r < 8; ++r) {
        const int gm = m0 + msub + mi * 16 + r + 8 * hi;
        const int b  = gm >> 11;
        const int s  = gm & (SEQ - 1);
        float val = a[r];
        if (which < 2) {
          float sn, cs;
          __sincosf((float)tok[s] * inv, &sn, &cs);
          const float partner = __shfl_xor(val, 1, 32);  // pair column
          val = (d & 1) ? fmaf(partner, sn, val * cs)    // odd:  e*sin + o*cos
                        : fmaf(val, cs, -partner * sn);  // even: e*cos - o*sin
          if (which == 0) val *= 0.125f;                 // fold 1/sqrt(DK)
        }
        dst[(((size_t)b * NH + head) * SEQ + s) * DK + d] = (_Float16)val;
      }
    }
  }
}

// =========================================================================
// Kernel 2: causal flash attention per (64-query tile, batch-head).
// Block 128 (4 waves); wave owns 16 query rows; online softmax over 64-wide
// key tiles; QK^T and PV via WMMA.  Q/K tile fills use async global->LDS
// copies when the toolchain exposes them.  Output f16 into [B][S][H*DK].
// =========================================================================
__global__ __launch_bounds__(128) void flash_attn_kernel(
    const _Float16* __restrict__ Qh, const _Float16* __restrict__ Kh,
    const _Float16* __restrict__ Vh, _Float16* __restrict__ An) {
  __shared__ _Float16 Qs[64 * 72];  // [q][d]
  __shared__ _Float16 Ks[64 * 72];  // [kpos][d]
  __shared__ _Float16 Vt[64 * 72];  // [d][kpos]  (transposed V)
  __shared__ _Float16 Ps[64 * 72];  // [q][kpos]  probs as f16

  const int tid  = threadIdx.x;
  const int lane = tid & 31;
  const int wave = tid >> 5;
  const int bh   = blockIdx.y;          // 0..31
  const int qb   = blockIdx.x;          // 0..31
  const size_t base = (size_t)bh * SEQ * DK;

  const int crow = tid >> 1, ccb = (tid & 1) * 32;  // cooperative copy coords

  {  // load 64x64 Q tile (f16 straight copy)
    const _Float16* src = Qh + base + (size_t)(qb * 64 + crow) * DK + ccb;
    _Float16* dst = Qs + crow * 72 + ccb;
#if HAVE_ASYNC
#pragma unroll
    for (int q = 0; q < 4; ++q) async_cp16(dst + q * 8, src + q * 8);
#else
#pragma unroll
    for (int q = 0; q < 4; ++q) ((v8h*)dst)[q] = ((const v8h*)src)[q];
#endif
  }
  ASYNC_WAIT();
  __syncthreads();

  const int qr0 = wave << 4;            // this wave's query-row base (local)
  const v16h qf0 = ld_a_frag(Qs, 72, qr0, 0);
  const v16h qf1 = ld_a_frag(Qs, 72, qr0, 32);

  const v8f zero = {};
  v8f O[4] = {zero, zero, zero, zero};
  float m_[8], l_[8];
#pragma unroll
  for (int r = 0; r < 8; ++r) { m_[r] = -INFINITY; l_[r] = 0.f; }

  const int hi = lane >> 4, ln = lane & 15;

  for (int kb = 0; kb <= qb; ++kb) {
    __syncthreads();  // protect Ks/Vt reuse
    {  // K tile -> LDS (straight)
      const _Float16* src = Kh + base + (size_t)(kb * 64 + crow) * DK + ccb;
      _Float16* dst = Ks + crow * 72 + ccb;
#if HAVE_ASYNC
#pragma unroll
      for (int q = 0; q < 4; ++q) async_cp16(dst + q * 8, src + q * 8);
#else
#pragma unroll
      for (int q = 0; q < 4; ++q) ((v8h*)dst)[q] = ((const v8h*)src)[q];
#endif
    }
    {  // V tile -> LDS transposed (scattered 16-bit stores)
      const v8h* src = (const v8h*)(Vh + base + (size_t)(kb * 64 + crow) * DK + ccb);
#pragma unroll
      for (int q = 0; q < 4; ++q) {
        v8h v = src[q];
#pragma unroll
        for (int j = 0; j < 8; ++j) Vt[(ccb + q * 8 + j) * 72 + crow] = v[j];
      }
    }
    if (kb < qb) {  // prefetch next K/V tiles (one cacheline per thread)
      const char* nk = (const char*)(Kh + base + (size_t)((kb + 1) * 64) * DK);
      const char* nv = (const char*)(Vh + base + (size_t)((kb + 1) * 64) * DK);
      __builtin_prefetch(nk + tid * 64, 0, 0);
      __builtin_prefetch(nv + tid * 64, 0, 0);
    }
    ASYNC_WAIT();
    __syncthreads();

    // scores: Q(16x64) @ K^T -> 4 n-tiles of 16 key positions
    v8f sc[4];
#pragma unroll
    for (int nt = 0; nt < 4; ++nt) {
      v16h b0 = ld_b_frag(Ks, 72, nt * 16, 0);
      v16h b1 = ld_b_frag(Ks, 72, nt * 16, 32);
      v8f s = zero;
      s = WMMA_F16(qf0, b0, s);
      s = WMMA_F16(qf1, b1, s);
      sc[nt] = s;
    }

    const bool diag = (kb == qb);
#pragma unroll
    for (int r = 0; r < 8; ++r) {
      const int qrow = qr0 + r + 8 * hi;        // local query row (0..63)
      float mx = m_[r];
#pragma unroll
      for (int nt = 0; nt < 4; ++nt) {
        float v = sc[nt][r];
        if (diag && (nt * 16 + ln) > qrow) { v = -INFINITY; sc[nt][r] = v; }
        mx = fmaxf(mx, v);
      }
#pragma unroll
      for (int off = 1; off < 16; off <<= 1)
        mx = fmaxf(mx, __shfl_xor(mx, off, 32));  // 16-lane row reduce
      const float scale = __expf(m_[r] - mx);
      float rs = 0.f;
#pragma unroll
      for (int nt = 0; nt < 4; ++nt) {
        const float p = __expf(sc[nt][r] - mx);
        sc[nt][r] = p;
        rs += p;
      }
#pragma unroll
      for (int off = 1; off < 16; off <<= 1) rs += __shfl_xor(rs, off, 32);
      l_[r] = l_[r] * scale + rs;
      m_[r] = mx;
#pragma unroll
      for (int dt = 0; dt < 4; ++dt) O[dt][r] *= scale;
    }

    // probs -> LDS f16 (this wave's 16x64 strip only; wave-local, in-order)
#pragma unroll
    for (int nt = 0; nt < 4; ++nt)
#pragma unroll
      for (int r = 0; r < 8; ++r)
        Ps[(qr0 + r + 8 * hi) * 72 + nt * 16 + ln] = (_Float16)sc[nt][r];

    // O += P(16x64) @ V(64x64)
    const v16h a0 = ld_a_frag(Ps, 72, qr0, 0);
    const v16h a1 = ld_a_frag(Ps, 72, qr0, 32);
#pragma unroll
    for (int dt = 0; dt < 4; ++dt) {
      v16h b0 = ld_b_frag(Vt, 72, dt * 16, 0);
      v16h b1 = ld_b_frag(Vt, 72, dt * 16, 32);
      O[dt] = WMMA_F16(a0, b0, O[dt]);
      O[dt] = WMMA_F16(a1, b1, O[dt]);
    }
  }

  // normalize + store into [B][S][D_MODEL] (head-interleaved) as f16
  const int b = bh >> 4, h = bh & 15;
#pragma unroll
  for (int r = 0; r < 8; ++r) {
    const float invl = 1.0f / l_[r];
    const int s = qb * 64 + qr0 + r + 8 * hi;
    const size_t off = ((size_t)b * SEQ + s) * D_MODEL + h * DK;
#pragma unroll
    for (int dt = 0; dt < 4; ++dt)
      An[off + dt * 16 + ln] = (_Float16)(O[dt][r] * invl);
  }
}

// =========================================================================
// Kernel 3: output projection  y[gm, n] = sum_v attn[gm, v] * wo[n, v]
// Same 128x64 / K-slab-64 tiling as kernel 1.  Output f32.
// =========================================================================
__global__ __launch_bounds__(256) void out_proj_kernel(
    const _Float16* __restrict__ An, const float* __restrict__ wo,
    float* __restrict__ out) {
  __shared__ _Float16 as_[128 * 72];
  __shared__ _Float16 bs_[64 * 72];

  const int tid  = threadIdx.x;
  const int lane = tid & 31;
  const int wave = tid >> 5;
  const int m0   = blockIdx.x * 128;
  const int n0   = blockIdx.y * 64;
  const int msub = (wave & 3) << 5;
  const int nsub = (wave >> 2) << 5;
  const int arow = tid >> 1, acb = (tid & 1) << 5;
  const int wrow = tid >> 2, wcb = (tid & 3) << 4;

  v8f acc[2][2] = {};

  for (int kk = 0; kk < D_MODEL; kk += 64) {
    __syncthreads();
    {  // A tile: already f16; async copy when available (32 halfs / thread)
      const _Float16* src = An + (size_t)(m0 + arow) * D_MODEL + kk + acb;
      _Float16* dst = as_ + arow * 72 + acb;
#if HAVE_ASYNC
#pragma unroll
      for (int q = 0; q < 4; ++q) async_cp16(dst + q * 8, src + q * 8);
#else
#pragma unroll
      for (int q = 0; q < 4; ++q) ((v8h*)dst)[q] = ((const v8h*)src)[q];
#endif
    }
    {  // wo tile: f32 -> f16
      const float* sw = wo + (size_t)(n0 + wrow) * D_MODEL + kk + wcb;
#pragma unroll
      for (int q = 0; q < 2; ++q)
        *(v8h*)(bs_ + wrow * 72 + wcb + q * 8) = cvt8(sw + q * 8);
    }
    ASYNC_WAIT();
    __syncthreads();

#pragma unroll
    for (int ks = 0; ks < 64; ks += 32) {
      v16h a0 = ld_a_frag(as_, 72, msub, ks);
      v16h a1 = ld_a_frag(as_, 72, msub + 16, ks);
      v16h b0 = ld_b_frag(bs_, 72, nsub, ks);
      v16h b1 = ld_b_frag(bs_, 72, nsub + 16, ks);
      acc[0][0] = WMMA_F16(a0, b0, acc[0][0]);
      acc[0][1] = WMMA_F16(a0, b1, acc[0][1]);
      acc[1][0] = WMMA_F16(a1, b0, acc[1][0]);
      acc[1][1] = WMMA_F16(a1, b1, acc[1][1]);
    }
  }

  const int hi = lane >> 4, ln = lane & 15;
#pragma unroll
  for (int mi = 0; mi < 2; ++mi) {
#pragma unroll
    for (int nb = 0; nb < 2; ++nb) {
      v8f a = acc[mi][nb];
      const int n = n0 + nsub + nb * 16 + ln;
#pragma unroll
      for (int r = 0; r < 8; ++r) {
        const int gm = m0 + msub + mi * 16 + r + 8 * hi;
        out[(size_t)gm * D_MODEL + n] = a[r];
      }
    }
  }
}

// =========================================================================
extern "C" void kernel_launch(void* const* d_in, const int* in_sizes, int n_in,
                              void* d_out, int out_size, void* d_ws,
                              size_t ws_size, hipStream_t stream) {
  (void)in_sizes; (void)n_in; (void)out_size; (void)ws_size;
  const float* x  = (const float*)d_in[0];
  const float* wq = (const float*)d_in[1];
  const float* wk = (const float*)d_in[2];
  const float* wv = (const float*)d_in[3];
  const float* wo = (const float*)d_in[4];
  const int*  tok = (const int*)d_in[5];

  const size_t per = (size_t)BATCH * NH * SEQ * DK;  // 4M halfs = 8MB
  _Float16* Qh = (_Float16*)d_ws;
  _Float16* Kh = Qh + per;
  _Float16* Vh = Kh + per;
  _Float16* An = Vh + per;  // total 32MB of workspace
  float* out = (float*)d_out;

  qkv_rope_kernel<<<dim3(BS / 128, NC / 64), 256, 0, stream>>>(
      x, wq, wk, wv, tok, Qh, Kh, Vh);
  flash_attn_kernel<<<dim3(SEQ / 64, BATCH * NH), 128, 0, stream>>>(
      Qh, Kh, Vh, An);
  out_proj_kernel<<<dim3(BS / 128, D_MODEL / 64), 256, 0, stream>>>(
      An, wo, out);
}
